// RotaryCrossAttention_6107443494995
// MI455X (gfx1250) — matrix operand order repaired and checked by reading, not compiled, not run on hardware
//
#include <hip/hip_runtime.h>
#include <hip/hip_bf16.h>

// Problem constants (match reference)
#define kB     4
#define kNQ    1024
#define kNKV   4096
#define kDim   512
#define kHeads 8
#define kDHead 64
#define kInner 512

typedef __attribute__((ext_vector_type(16))) __bf16 v16bf;
typedef __attribute__((ext_vector_type(8)))  float  v8f;
typedef __attribute__((ext_vector_type(4)))  unsigned int v4u;

// 32-byte fragment viewed either as a WMMA operand or two 16B vector loads.
union ABf { v16bf v; v4u u[2]; };

static __device__ __forceinline__ v8f wmma_bf16(v16bf a, v16bf b, v8f c) {
  // D = A(16x32 bf16) * B(32x16 bf16) + C(16x16 f32)
  return __builtin_amdgcn_wmma_f32_16x16x32_bf16(false, a, false, b, (short)0, c,
                                                 false, false);
}

// ---------------- LayerNorm (fp32 in, bf16 out), row width 512 --------------
__global__ __launch_bounds__(128)
void ln512_bf16(const float* __restrict__ x, const float* __restrict__ g,
                const float* __restrict__ beta, __bf16* __restrict__ y) {
  const int row = blockIdx.x;
  const float4* xr = (const float4*)(x + (size_t)row * kDim);
  float4 v = xr[threadIdx.x];
  float s  = v.x + v.y + v.z + v.w;
  float s2 = v.x * v.x + v.y * v.y + v.z * v.z + v.w * v.w;
#pragma unroll
  for (int m = 16; m >= 1; m >>= 1) {
    s  += __shfl_xor(s, m);
    s2 += __shfl_xor(s2, m);
  }
  __shared__ float ls[4], ls2[4];
  const int w = threadIdx.x >> 5;
  if ((threadIdx.x & 31) == 0) { ls[w] = s; ls2[w] = s2; }
  __syncthreads();
  s  = ls[0] + ls[1] + ls[2] + ls[3];
  s2 = ls2[0] + ls2[1] + ls2[2] + ls2[3];
  const float mean = s * (1.0f / kDim);
  const float var  = s2 * (1.0f / kDim) - mean * mean;
  const float inv  = rsqrtf(var + 1e-5f);
  __bf16* yr = y + (size_t)row * kDim;
  const int c0 = threadIdx.x * 4;
  const float xv[4] = {v.x, v.y, v.z, v.w};
#pragma unroll
  for (int i = 0; i < 4; ++i)
    yr[c0 + i] = (__bf16)((xv[i] - mean) * inv * g[c0 + i] + beta[c0 + i]);
}

// ------------- Weight convert: fp32 [K][Nin] slice -> bf16 [N][K] -----------
__global__ __launch_bounds__(256)
void transpose_cvt(const float* __restrict__ in, __bf16* __restrict__ out,
                   int K, int Nin, int col0, int N) {
  int idx = blockIdx.x * 256 + threadIdx.x;
  if (idx >= K * N) return;
  const int k = idx % K;
  const int n = idx / K;
  out[(size_t)n * K + k] = (__bf16)in[(size_t)k * Nin + col0 + n];
}

// -- bf16 TN GEMM, 64x64 register-blocked: C[M][N](+bias) = A[M][K]*Bt[N][K]^T
// One wave owns a 64x64 output macro-tile (4x4 grid of 16x16 WMMA tiles).
// Per K-step of 32: 4 A-fragments + 4 B-fragments feed 16 WMMAs -> 4x operand
// reuse vs. the naive 1-tile/wave version, moving the kernel from L2-load-
// bound toward the WMMA math ceiling. Fragment loads use a matching K order
// for both operands (A: 16 contiguous K at half*16 for row lane&15; B: 16
// contiguous K at half*16 for column lane&15).
__global__ __launch_bounds__(32)
void gemm_bf16_64x64(const __bf16* __restrict__ A, const __bf16* __restrict__ Bt,
                     float* __restrict__ C, const float* __restrict__ bias,
                     int M, int N, int K) {
  const int lane = threadIdx.x;
  const int half = lane >> 4, n16 = lane & 15;
  const int m0 = blockIdx.x * 64, n0 = blockIdx.y * 64;

  const __bf16* arow[4];
  const __bf16* brow[4];
#pragma unroll
  for (int i = 0; i < 4; ++i) {
    arow[i] = A  + (size_t)(m0 + i * 16 + n16) * K;
    brow[i] = Bt + (size_t)(n0 + i * 16 + n16) * K;
  }

  v8f acc[4][4];
#pragma unroll
  for (int i = 0; i < 4; ++i)
#pragma unroll
    for (int j = 0; j < 4; ++j)
#pragma unroll
      for (int r = 0; r < 8; ++r) acc[i][j][r] = 0.0f;

  for (int k0 = 0; k0 < K; k0 += 32) {
    ABf a[4], b[4];
#pragma unroll
    for (int i = 0; i < 4; ++i) {
      __builtin_prefetch(arow[i] + k0 + 128, 0, 1);  // global_prefetch_b8
      a[i].u[0] = *(const v4u*)(arow[i] + k0 + half * 16);
      a[i].u[1] = *(const v4u*)(arow[i] + k0 + half * 16 + 8);
    }
#pragma unroll
    for (int j = 0; j < 4; ++j) {
      __builtin_prefetch(brow[j] + k0 + 128, 0, 1);
      b[j].u[0] = *(const v4u*)(brow[j] + k0 + half * 16);
      b[j].u[1] = *(const v4u*)(brow[j] + k0 + half * 16 + 8);
    }
#pragma unroll
    for (int i = 0; i < 4; ++i)
#pragma unroll
      for (int j = 0; j < 4; ++j)
        acc[i][j] = wmma_bf16(a[i].v, b[j].v, acc[i][j]);
  }

#pragma unroll
  for (int j = 0; j < 4; ++j) {
    const int n = n0 + j * 16 + n16;
    const float bv = bias ? bias[n] : 0.0f;
#pragma unroll
    for (int i = 0; i < 4; ++i) {
#pragma unroll
      for (int r = 0; r < 8; ++r) {
        const int m = m0 + i * 16 + r + half * 8;  // C/D layout: VGPR r -> row r (+8 upper half)
        C[(size_t)m * N + n] = acc[i][j][r] + bv;
      }
    }
  }
}

// ---- RoPE + head pack: fp32 [B*N][INNER] -> bf16 [B][H][N][64] (scaled) ----
__global__ __launch_bounds__(256)
void rope_pack(const float* __restrict__ x, const float* __restrict__ pos,
               __bf16* __restrict__ out, int N, float scale) {
  const int idx = blockIdx.x * 256 + threadIdx.x;       // one rotation pair
  if (idx >= kB * N * (kInner / 2)) return;
  const int d2 = idx & 31;
  int t = idx >> 5;
  const int h = t & 7; t >>= 3;
  const int n = t % N, b = t / N;
  const int d0 = d2 * 2;
  const float* xr = x + ((size_t)(b * N + n)) * kInner + h * kDHead;
  const float* pr = pos + ((size_t)(b * N + n)) * kDHead;
  const float x0 = xr[d0], x1 = xr[d0 + 1];
  const float c0 = __cosf(pr[d0]),     s0 = __sinf(pr[d0]);
  const float c1 = __cosf(pr[d0 + 1]), s1 = __sinf(pr[d0 + 1]);
  __bf16* orow = out + ((size_t)((b * kHeads + h) * N + n)) * kDHead;
  orow[d0]     = (__bf16)((x0 * c0 - x1 * s0) * scale);
  orow[d0 + 1] = (__bf16)((x1 * c1 + x0 * s1) * scale);
}

// ---- RoPE + transposed pack for V: fp32 [B*NKV][INNER] -> bf16 [B][H][64][NKV]
__global__ __launch_bounds__(256)
void rope_pack_vT(const float* __restrict__ x, const float* __restrict__ pos,
                  __bf16* __restrict__ out) {
  const int idx = blockIdx.x * 256 + threadIdx.x;
  if (idx >= kB * kNKV * (kInner / 2)) return;
  const int d2 = idx & 31;
  int t = idx >> 5;
  const int h = t & 7; t >>= 3;
  const int n = t % kNKV, b = t / kNKV;
  const int d0 = d2 * 2;
  const float* xr = x + ((size_t)(b * kNKV + n)) * kInner + h * kDHead;
  const float* pr = pos + ((size_t)(b * kNKV + n)) * kDHead;
  const float x0 = xr[d0], x1 = xr[d0 + 1];
  const float c0 = __cosf(pr[d0]),     s0 = __sinf(pr[d0]);
  const float c1 = __cosf(pr[d0 + 1]), s1 = __sinf(pr[d0 + 1]);
  __bf16* obase = out + (size_t)(b * kHeads + h) * kDHead * kNKV;
  obase[(size_t)d0 * kNKV + n]       = (__bf16)(x0 * c0 - x1 * s0);
  obase[(size_t)(d0 + 1) * kNKV + n] = (__bf16)(x1 * c1 + x0 * s1);
}

// ---- Inverse RoPE (cos p, -sin p) + repack: fp32 [B][H][NQ][64] -> bf16 [B*NQ][INNER]
__global__ __launch_bounds__(256)
void rope_inv_pack(const float* __restrict__ x, const float* __restrict__ pos,
                   __bf16* __restrict__ out) {
  const int idx = blockIdx.x * 256 + threadIdx.x;
  if (idx >= kB * kNQ * (kInner / 2)) return;
  const int d2 = idx & 31;
  int t = idx >> 5;
  const int h = t & 7; t >>= 3;
  const int n = t % kNQ, b = t / kNQ;
  const int d0 = d2 * 2;
  const float* xr = x + ((size_t)((b * kHeads + h) * kNQ + n)) * kDHead;
  const float* pr = pos + ((size_t)(b * kNQ + n)) * kDHead;
  const float x0 = xr[d0], x1 = xr[d0 + 1];
  const float c0 = __cosf(pr[d0]),     s0 = __sinf(pr[d0]);
  const float c1 = __cosf(pr[d0 + 1]), s1 = __sinf(pr[d0 + 1]);
  __bf16* orow = out + ((size_t)(b * kNQ + n)) * kInner + h * kDHead;
  orow[d0]     = (__bf16)(x0 * c0 + x1 * s0);
  orow[d0 + 1] = (__bf16)(x1 * c1 - x0 * s1);
}

// --------------------- Flash attention (one wave / q-tile) ------------------
// Q: bf16 [B][H][NQ][64] (pre-scaled by 1/8), K: bf16 [B][H][NKV][64],
// Vt: bf16 [B][H][64][NKV], mask: int [B][NKV], O: fp32 [B][H][NQ][64].
__global__ __launch_bounds__(32)
void flash_attn(const __bf16* __restrict__ Q, const __bf16* __restrict__ Kh,
                const __bf16* __restrict__ Vt, const int* __restrict__ mask,
                float* __restrict__ O) {
  const int lane = threadIdx.x;
  const int half = lane >> 4, n16 = lane & 15;
  const int q0 = blockIdx.x * 16;
  const size_t bh = (size_t)(blockIdx.z * kHeads + blockIdx.y);

  // Preload both 16x32 A-fragments of the q tile (d = 0..31, 32..63).
  const __bf16* qrow = Q + (bh * kNQ + q0 + n16) * kDHead;
  ABf aq0, aq1;
  aq0.u[0] = *(const v4u*)(qrow + half * 16);
  aq0.u[1] = *(const v4u*)(qrow + half * 16 + 8);
  aq1.u[0] = *(const v4u*)(qrow + 32 + half * 16);
  aq1.u[1] = *(const v4u*)(qrow + 32 + half * 16 + 8);

  __shared__ float sP[16][32];  // C-layout -> A-layout staging for P

  float mrow[8], lrow[8];
  v8f acc[4];
#pragma unroll
  for (int j = 0; j < 8; ++j) {
    mrow[j] = -1e30f;
    lrow[j] = 0.0f;
#pragma unroll
    for (int dt = 0; dt < 4; ++dt) acc[dt][j] = 0.0f;
  }
  const int* mcol = mask + (size_t)blockIdx.z * kNKV;

  for (int kv0 = 0; kv0 < kNKV; kv0 += 32) {
    // S tiles: scores for 16 q rows x 32 kv columns
    v8f s[2];
#pragma unroll
    for (int t = 0; t < 2; ++t) {
      const __bf16* krow = Kh + (bh * kNKV + kv0 + t * 16 + n16) * kDHead;
      ABf bk0, bk1;
      bk0.u[0] = *(const v4u*)(krow + half * 16);
      bk0.u[1] = *(const v4u*)(krow + half * 16 + 8);
      bk1.u[0] = *(const v4u*)(krow + 32 + half * 16);
      bk1.u[1] = *(const v4u*)(krow + 32 + half * 16 + 8);
      v8f c;
#pragma unroll
      for (int j = 0; j < 8; ++j) c[j] = 0.0f;
      c = wmma_bf16(aq0.v, bk0.v, c);
      c = wmma_bf16(aq1.v, bk1.v, c);
      s[t] = c;
    }
    // Mask (each lane owns one kv column per tile)
#pragma unroll
    for (int t = 0; t < 2; ++t)
      if (mcol[kv0 + t * 16 + n16] == 0) {
#pragma unroll
        for (int j = 0; j < 8; ++j) s[t][j] = -1e30f;
      }
    // Online softmax: row reductions across the 16 lanes of each half
    float rmax[8], rsum[8], alpha[8];
#pragma unroll
    for (int j = 0; j < 8; ++j) rmax[j] = fmaxf(s[0][j], s[1][j]);
#pragma unroll
    for (int m = 8; m >= 1; m >>= 1)
#pragma unroll
      for (int j = 0; j < 8; ++j) rmax[j] = fmaxf(rmax[j], __shfl_xor(rmax[j], m));
#pragma unroll
    for (int j = 0; j < 8; ++j) {
      const float mn = fmaxf(mrow[j], rmax[j]);
      alpha[j] = __expf(mrow[j] - mn);
      mrow[j]  = mn;
      s[0][j]  = __expf(s[0][j] - mn);
      s[1][j]  = __expf(s[1][j] - mn);
      rsum[j]  = s[0][j] + s[1][j];
    }
#pragma unroll
    for (int m = 8; m >= 1; m >>= 1)
#pragma unroll
      for (int j = 0; j < 8; ++j) rsum[j] += __shfl_xor(rsum[j], m);
#pragma unroll
    for (int j = 0; j < 8; ++j) {
      lrow[j] = lrow[j] * alpha[j] + rsum[j];
#pragma unroll
      for (int dt = 0; dt < 4; ++dt) acc[dt][j] *= alpha[j];
    }
    // P: C-layout -> LDS -> A-layout (16x32), then acc += P @ V
#pragma unroll
    for (int j = 0; j < 8; ++j) {
      sP[j + half * 8][n16]      = s[0][j];
      sP[j + half * 8][16 + n16] = s[1][j];
    }
    v16bf ap;
#pragma unroll
    for (int e = 0; e < 8; ++e) ap[e] = (__bf16)sP[n16][half * 16 + e];
#pragma unroll
    for (int e = 0; e < 8; ++e) ap[8 + e] = (__bf16)sP[n16][half * 16 + 8 + e];
#pragma unroll
    for (int dt = 0; dt < 4; ++dt) {
      const __bf16* vrow = Vt + (bh * kDHead + dt * 16 + n16) * kNKV + kv0 + half * 16;
      ABf bv;
      bv.u[0] = *(const v4u*)(vrow);
      bv.u[1] = *(const v4u*)(vrow + 8);
      acc[dt] = wmma_bf16(ap, bv.v, acc[dt]);
    }
  }
  // Normalize and store
#pragma unroll
  for (int j = 0; j < 8; ++j) {
    const float invl = 1.0f / lrow[j];
    float* orow = O + (bh * kNQ + q0 + j + half * 8) * kDHead;
#pragma unroll
    for (int dt = 0; dt < 4; ++dt) orow[dt * 16 + n16] = acc[dt][j] * invl;
  }
}

// ----------------------------------------------------------------------------
extern "C" void kernel_launch(void* const* d_in, const int* in_sizes, int n_in,
                              void* d_out, int out_size, void* d_ws, size_t ws_size,
                              hipStream_t stream) {
  (void)in_sizes; (void)n_in; (void)out_size; (void)ws_size;
  const float* x_query = (const float*)d_in[0];
  const float* x_ctx   = (const float*)d_in[1];
  const float* qpos    = (const float*)d_in[2];
  const float* cpos    = (const float*)d_in[3];
  const int*   cmask   = (const int*)d_in[4];
  const float* ln_q_w  = (const float*)d_in[5];
  const float* ln_q_b  = (const float*)d_in[6];
  const float* ln_c_w  = (const float*)d_in[7];
  const float* ln_c_b  = (const float*)d_in[8];
  const float* w_q     = (const float*)d_in[9];
  const float* w_kv    = (const float*)d_in[10];
  const float* w_out   = (const float*)d_in[11];
  const float* b_out   = (const float*)d_in[12];

  char* ws = (char*)d_ws;
  size_t off = 0;
  auto alloc = [&](size_t bytes) -> void* {
    void* p = ws + off;
    off = (off + bytes + 255) & ~(size_t)255;
    return p;
  };
  __bf16* xq_bf = (__bf16*)alloc((size_t)kB * kNQ * kDim * 2);
  __bf16* xc_bf = (__bf16*)alloc((size_t)kB * kNKV * kDim * 2);
  __bf16* wq_t  = (__bf16*)alloc((size_t)kDim * kInner * 2);
  __bf16* wk_t  = (__bf16*)alloc((size_t)kDim * kInner * 2);
  __bf16* wv_t  = (__bf16*)alloc((size_t)kDim * kInner * 2);
  __bf16* wo_t  = (__bf16*)alloc((size_t)kInner * kDim * 2);
  __bf16* q_hd  = (__bf16*)alloc((size_t)kB * kHeads * kNQ * kDHead * 2);
  __bf16* k_hd  = (__bf16*)alloc((size_t)kB * kHeads * kNKV * kDHead * 2);
  __bf16* vT_hd = (__bf16*)alloc((size_t)kB * kHeads * kDHead * kNKV * 2);
  __bf16* at_bf = (__bf16*)alloc((size_t)kB * kNQ * kInner * 2);
  // 32 MiB scratch reused sequentially: q_f32 -> k_f32 -> v_f32 -> attnO
  float* scratch = (float*)alloc((size_t)kB * kNKV * kInner * 4);

  // 1) LayerNorms -> bf16
  ln512_bf16<<<kB * kNQ, 128, 0, stream>>>(x_query, ln_q_w, ln_q_b, xq_bf);
  ln512_bf16<<<kB * kNKV, 128, 0, stream>>>(x_ctx, ln_c_w, ln_c_b, xc_bf);

  // 2) Weights -> bf16 [N][K]
  const int wElems = kDim * kInner;
  transpose_cvt<<<(wElems + 255) / 256, 256, 0, stream>>>(w_q, wq_t, kDim, kInner, 0, kInner);
  transpose_cvt<<<(wElems + 255) / 256, 256, 0, stream>>>(w_kv, wk_t, kDim, 2 * kInner, 0, kInner);
  transpose_cvt<<<(wElems + 255) / 256, 256, 0, stream>>>(w_kv, wv_t, kDim, 2 * kInner, kInner, kInner);
  transpose_cvt<<<(wElems + 255) / 256, 256, 0, stream>>>(w_out, wo_t, kInner, kDim, 0, kDim);

  // 3) q = LN(xq) @ w_q ; RoPE (+1/sqrt(64) scale) ; pack heads
  {
    dim3 g(kB * kNQ / 64, kInner / 64);
    gemm_bf16_64x64<<<g, 32, 0, stream>>>(xq_bf, wq_t, scratch, nullptr,
                                          kB * kNQ, kInner, kDim);
    const int tot = kB * kNQ * (kInner / 2);
    rope_pack<<<(tot + 255) / 256, 256, 0, stream>>>(scratch, qpos, q_hd, kNQ, 0.125f);
  }
  // 4) k = LN(xc) @ w_k ; RoPE ; pack heads
  {
    dim3 g(kB * kNKV / 64, kInner / 64);
    gemm_bf16_64x64<<<g, 32, 0, stream>>>(xc_bf, wk_t, scratch, nullptr,
                                          kB * kNKV, kInner, kDim);
    const int tot = kB * kNKV * (kInner / 2);
    rope_pack<<<(tot + 255) / 256, 256, 0, stream>>>(scratch, cpos, k_hd, kNKV, 1.0f);
    // 5) v = LN(xc) @ w_v ; RoPE ; pack transposed
    gemm_bf16_64x64<<<g, 32, 0, stream>>>(xc_bf, wv_t, scratch, nullptr,
                                          kB * kNKV, kInner, kDim);
    rope_pack_vT<<<(tot + 255) / 256, 256, 0, stream>>>(scratch, cpos, vT_hd);
  }
  // 6) Flash attention -> fp32 [B][H][NQ][64] (reuse scratch)
  {
    dim3 g(kNQ / 16, kHeads, kB);
    flash_attn<<<g, 32, 0, stream>>>(q_hd, k_hd, vT_hd, cmask, scratch);
  }
  // 7) Inverse RoPE + repack to [B*NQ][INNER] bf16
  {
    const int tot = kB * kNQ * (kInner / 2);
    rope_inv_pack<<<(tot + 255) / 256, 256, 0, stream>>>(scratch, qpos, at_bf);
  }
  // 8) Final projection + bias -> d_out (fp32)
  {
    dim3 g(kB * kNQ / 64, kDim / 64);
    gemm_bf16_64x64<<<g, 32, 0, stream>>>(at_bf, wo_t, (float*)d_out, b_out,
                                          kB * kNQ, kDim, kInner);
  }
}
